// AdaptiveRotatedScaledConv2d_33938831573549
// MI455X (gfx1250) — matrix operand order repaired and checked by reading, not compiled, not run on hardware
//
#include <hip/hip_runtime.h>

typedef __bf16 v16bf __attribute__((ext_vector_type(16)));
typedef __bf16 v8bf  __attribute__((ext_vector_type(8)));
typedef float  v8f   __attribute__((ext_vector_type(8)));

#define B_    16
#define CIN   256
#define COUT  256
#define HH    64
#define WW    64
#define NK    4
#define KTOT  (CIN * 9)     // 2304
#define HWSZ  (HH * WW)     // 4096
#define LDA   48            // LDS row stride (bf16) -> 96 bytes

// ---------------------------------------------------------------------------
// Kernel 1: build rot[b][g][9][9] * lambda  (64 tiny matrices)
// ---------------------------------------------------------------------------
__global__ void rot_kernel(const float* __restrict__ thetas,
                           const float* __restrict__ scales,
                           const float* __restrict__ lambdas,
                           float* __restrict__ rot) {
  int id = blockIdx.x * blockDim.x + threadIdx.x;
  if (id >= B_ * NK) return;
  float t = thetas[id], s = scales[id], lam = lambdas[id];
  float x = __cosf(t) * s, y = __sinf(t) * s, yp = -y;
  float a = x - y, b = x * y, c = x + y, d = a * c, e = a + c;
  float ap = x - yp, bp = x * yp, cp = x + yp, dp = ap * cp, ep = ap + cp;
  float M[81];
#pragma unroll
  for (int i = 0; i < 81; ++i) M[i] = 0.f;
  M[4 * 9 + 4] = 1.f;
  bool pos = (t >= 0.f);
  bool big = (s >= 1.f);
  bool m1  = (fabsf(t) <= 0.7853981633974483f);
  if (pos) {
    if (big && m1) {            // pb1
      M[0]=a;        M[1]=1.f-a;
      M[9+1]=1.f-y;  M[9+2]=y;
      M[18+2]=a;     M[18+5]=1.f-a;
      M[27+0]=y;     M[27+3]=1.f-y;
      M[45+5]=1.f-y; M[45+8]=y;
      M[54+3]=1.f-a; M[54+6]=a;
      M[63+6]=y;     M[63+7]=1.f-y;
      M[72+7]=1.f-a; M[72+8]=a;
    } else if (!big && m1) {    // ps1
      M[0]=d;  M[1]=a-d;  M[3]=c-d;  M[4]=1.f-e+d;
      M[9+1]=x-b;  M[9+2]=b;  M[9+4]=1.f-c+b;  M[9+5]=y-b;
      M[18+1]=c-d; M[18+2]=d; M[18+4]=1.f-e+d; M[18+5]=a-d;
      M[27+0]=b;   M[27+1]=y-b; M[27+3]=x-b;   M[27+4]=1.f-c+b;
      M[45+4]=1.f-c+b; M[45+5]=x-b; M[45+7]=y-b; M[45+8]=b;
      M[54+3]=a-d; M[54+4]=1.f-e+d; M[54+6]=d;  M[54+7]=c-d;
      M[63+3]=y-b; M[63+4]=1.f-c+b; M[63+6]=b;  M[63+7]=x-b;
      M[72+4]=1.f-e+d; M[72+5]=c-d; M[72+7]=a-d; M[72+8]=d;
    } else {                    // pb2 == ps2
      M[0]=a;  M[1]=1.f-a;
      M[9+1]=x-b;  M[9+2]=b;  M[9+4]=1.f-c+b;  M[9+5]=y-b;
      M[18+2]=a;   M[18+5]=1.f-a;
      M[27+0]=b;   M[27+1]=y-b; M[27+3]=x-b;   M[27+4]=1.f-c+b;
      M[45+4]=1.f-c+b; M[45+5]=x-b; M[45+7]=y-b; M[45+8]=b;
      M[54+3]=1.f-a; M[54+6]=a;
      M[63+3]=y-b; M[63+4]=1.f-c+b; M[63+6]=b;  M[63+7]=x-b;
      M[72+7]=1.f-a; M[72+8]=a;
    }
  } else {
    if (big && m1) {            // nb1
      M[0]=cp;        M[3]=1.f-cp;
      M[9+0]=yp;      M[9+1]=1.f-yp;
      M[18+1]=1.f-cp; M[18+2]=cp;
      M[27+3]=1.f-yp; M[27+6]=yp;
      M[45+2]=yp;     M[45+5]=1.f-yp;
      M[54+6]=cp;     M[54+7]=1.f-cp;
      M[63+7]=1.f-yp; M[63+8]=yp;
      M[72+5]=1.f-cp; M[72+8]=cp;
    } else if (!big && m1) {    // ns1
      M[0]=dp; M[1]=cp-dp; M[3]=ap-dp; M[4]=1.f-ep+dp;
      M[9+0]=bp;  M[9+1]=x-bp;  M[9+3]=yp-bp;  M[9+4]=1.f-cp+bp;
      M[18+1]=ap-dp; M[18+2]=dp; M[18+4]=1.f-ep+dp; M[18+5]=cp-dp;
      M[27+1]=yp-bp; M[27+2]=bp; M[27+4]=1.f-cp+bp; M[27+5]=x-bp;
      M[45+3]=x-bp;  M[45+4]=1.f-cp+bp; M[45+6]=bp; M[45+7]=yp-bp;
      M[54+3]=cp-dp; M[54+4]=1.f-ep+dp; M[54+6]=dp; M[54+7]=ap-dp;
      M[63+4]=1.f-cp+bp; M[63+5]=yp-bp; M[63+7]=x-bp; M[63+8]=bp;
      M[72+4]=1.f-ep+dp; M[72+5]=ap-dp; M[72+7]=cp-dp; M[72+8]=dp;
    } else {                    // nb2 == ns2
      M[0]=cp; M[3]=1.f-cp;
      M[9+0]=bp;  M[9+1]=x-bp;  M[9+3]=yp-bp;  M[9+4]=1.f-cp+bp;
      M[18+1]=1.f-cp; M[18+2]=cp;
      M[27+3]=x-bp;  M[27+4]=1.f-cp+bp; M[27+6]=bp; M[27+7]=yp-bp;
      M[45+1]=yp-bp; M[45+2]=bp; M[45+4]=1.f-cp+bp; M[45+5]=x-bp;
      M[54+6]=cp;    M[54+7]=1.f-cp;
      M[63+4]=1.f-cp+bp; M[63+5]=yp-bp; M[63+7]=x-bp; M[63+8]=bp;
      M[72+5]=1.f-cp; M[72+8]=cp;
    }
  }
#pragma unroll
  for (int i = 0; i < 81; ++i) rot[id * 81 + i] = M[i] * lam;
}

// ---------------------------------------------------------------------------
// Kernel 2: per-batch transformed weights, bf16, layout gw[b][o][c*9+tap]
// ---------------------------------------------------------------------------
__global__ __launch_bounds__(256) void wtrans_kernel(
    const float* __restrict__ weight, const float* __restrict__ rot,
    __bf16* __restrict__ gw) {
  __shared__ float rsh[NK * 81];
  const int b = blockIdx.x >> 8;
  const int blk = blockIdx.x & 255;
  for (int i = threadIdx.x; i < NK * 81; i += 256)
    rsh[i] = rot[b * NK * 81 + i];
  __syncthreads();

  const int lid = blk * 256 + (int)threadIdx.x;   // 0 .. 65535
  const int o = lid >> 8, c = lid & 255;
  float wj[NK][9];
#pragma unroll
  for (int n = 0; n < NK; ++n)
#pragma unroll
    for (int j = 0; j < 9; ++j)
      wj[n][j] = weight[((size_t)(n * COUT + o) * CIN + c) * 9 + j];

  __bf16* dst = gw + (size_t)(b * COUT + o) * KTOT + c * 9;
#pragma unroll
  for (int i = 0; i < 9; ++i) {
    float acc = 0.f;
#pragma unroll
    for (int n = 0; n < NK; ++n)
#pragma unroll
      for (int j = 0; j < 9; ++j)
        acc = fmaf(rsh[n * 81 + i * 9 + j], wj[n][j], acc);
    dst[i] = (__bf16)acc;
  }
}

// ---------------------------------------------------------------------------
// Kernel 3: implicit-GEMM conv via v_wmma_f32_16x16x32_bf16
//   block tile: 256 (Cout) x 32 (spatial); 8 waves, each 32M x 32N
//   -> 4 WMMAs / wave / K-step; double-buffered LDS, one barrier per step;
//   A tile streamed one step ahead via global_load_async_to_lds_b128.
// ---------------------------------------------------------------------------
__global__ __launch_bounds__(256) void conv_wmma_kernel(
    const float* __restrict__ x, const __bf16* __restrict__ gw,
    float* __restrict__ out) {
  __shared__ __bf16 As[2][256][LDA];   // 2 x 24576 B
  __shared__ __bf16 Bs[2][32][LDA];    // 2 x  3072 B

  const int b     = blockIdx.z;
  const int nBase = blockIdx.x * 32;
  const int tid   = threadIdx.x;
  const int lane  = tid & 31;
  const int wave  = tid >> 5;
  const int kHalf = lane >> 4;   // which K-half this lane carries
  const int lr    = lane & 15;

  v8f acc00 = {}, acc01 = {}, acc10 = {}, acc11 = {};

  // --- A async-stream state: thread t owns weight row t, 64 B per K-step ---
  const __bf16* gwB = gw + (size_t)b * COUT * KTOT;
  unsigned long long gaddr =
      (unsigned long long)(const void*)(gwB + (size_t)tid * KTOT);
  // flat shared address: low 32 bits are the LDS byte offset
  const unsigned ldsA0 = (unsigned)(size_t)(void*)&As[0][tid][0];
  const unsigned aBufBytes = (unsigned)sizeof(As[0]);   // 24576

  // IOFFSET is added to BOTH the LDS and global address (async-copy spec),
  // so one LDS-offset VGPR + one 64-bit VGPR pair covers all 4 b128 chunks.
  auto issueA = [&](int buf) {
    unsigned lds = ldsA0 + (unsigned)buf * aBufBytes;
    asm volatile(
        "global_load_async_to_lds_b128 %0, %1, off\n\t"
        "global_load_async_to_lds_b128 %0, %1, off offset:16\n\t"
        "global_load_async_to_lds_b128 %0, %1, off offset:32\n\t"
        "global_load_async_to_lds_b128 %0, %1, off offset:48"
        :: "v"(lds), "v"(gaddr) : "memory");
    gaddr += 64;
  };

  // --- B im2col incremental state: (c, tap) advance by +32 in K per step ---
  int cc[4], tp[4], hb[4], wb[4];
#pragma unroll
  for (int i = 0; i < 4; ++i) {
    const int e  = tid + i * 256;
    const int kl = e >> 5;
    const int nn = e & 31;
    cc[i] = kl / 9;
    tp[i] = kl - cc[i] * 9;
    const int p = nBase + nn;
    hb[i] = p >> 6;
    wb[i] = p & 63;
  }
  const float* xb = x + ((size_t)b * CIN << 12);

  auto gatherB = [&](int buf) {
#pragma unroll
    for (int i = 0; i < 4; ++i) {
      const int e  = tid + i * 256;
      const int kl = e >> 5;
      const int nn = e & 31;
      const int t3 = (tp[i] >= 6) ? 2 : ((tp[i] >= 3) ? 1 : 0);
      const int h  = hb[i] + t3 - 1;
      const int w  = wb[i] + (tp[i] - 3 * t3) - 1;
      float v = 0.f;
      if ((unsigned)h < (unsigned)HH && (unsigned)w < (unsigned)WW)
        v = xb[((size_t)cc[i] << 12) + (h << 6) + w];
      Bs[buf][nn][kl] = (__bf16)v;
      tp[i] += 5; cc[i] += 3;
      if (tp[i] >= 9) { tp[i] -= 9; cc[i] += 1; }
    }
  };

  // --- prologue: stage K-step 0 into buffer 0 ---
  issueA(0);
  gatherB(0);
  asm volatile("s_wait_asynccnt 0x0" ::: "memory");
  __syncthreads();

  int cur = 0;
  for (int k0 = 0; k0 < KTOT; k0 += 32) {
    const int nxt = cur ^ 1;
    const bool more = (k0 + 32 < KTOT);
    if (more) {          // stage next step while WMMAs consume current
      issueA(nxt);
      gatherB(nxt);
    }

    // --- fragments per documented wave32 VGPR layouts ---
    const int r0 = wave * 32 + lr;
    v8bf a0lo = *(const v8bf*)&As[cur][r0][kHalf * 8];
    v8bf a0hi = *(const v8bf*)&As[cur][r0][kHalf * 8 + 16];
    v8bf a1lo = *(const v8bf*)&As[cur][r0 + 16][kHalf * 8];
    v8bf a1hi = *(const v8bf*)&As[cur][r0 + 16][kHalf * 8 + 16];
    v8bf b0lo = *(const v8bf*)&Bs[cur][lr][kHalf * 8];
    v8bf b0hi = *(const v8bf*)&Bs[cur][lr][kHalf * 8 + 16];
    v8bf b1lo = *(const v8bf*)&Bs[cur][16 + lr][kHalf * 8];
    v8bf b1hi = *(const v8bf*)&Bs[cur][16 + lr][kHalf * 8 + 16];
    v16bf a0 = __builtin_shufflevector(a0lo, a0hi, 0,1,2,3,4,5,6,7,8,9,10,11,12,13,14,15);
    v16bf a1 = __builtin_shufflevector(a1lo, a1hi, 0,1,2,3,4,5,6,7,8,9,10,11,12,13,14,15);
    v16bf b0 = __builtin_shufflevector(b0lo, b0hi, 0,1,2,3,4,5,6,7,8,9,10,11,12,13,14,15);
    v16bf b1 = __builtin_shufflevector(b1lo, b1hi, 0,1,2,3,4,5,6,7,8,9,10,11,12,13,14,15);

    acc00 = __builtin_amdgcn_wmma_f32_16x16x32_bf16(false, a0, false, b0,
                                                    (short)0, acc00, false, false);
    acc01 = __builtin_amdgcn_wmma_f32_16x16x32_bf16(false, a0, false, b1,
                                                    (short)0, acc01, false, false);
    acc10 = __builtin_amdgcn_wmma_f32_16x16x32_bf16(false, a1, false, b0,
                                                    (short)0, acc10, false, false);
    acc11 = __builtin_amdgcn_wmma_f32_16x16x32_bf16(false, a1, false, b1,
                                                    (short)0, acc11, false, false);

    if (more) {
      asm volatile("s_wait_asynccnt 0x0" ::: "memory");
      __syncthreads();   // next buffers ready; current fully consumed
    }
    cur = nxt;
  }

  // --- epilogue: C layout lane half -> M+8, VGPR r -> row r ---
  const int mg = wave * 32 + kHalf * 8;
  float* o0 = out + (size_t)b * COUT * HWSZ + nBase;
#pragma unroll
  for (int r = 0; r < 8; ++r) {
    o0[(size_t)(mg + r) * HWSZ + lr]           = acc00[r];
    o0[(size_t)(mg + r) * HWSZ + 16 + lr]      = acc01[r];
    o0[(size_t)(mg + 16 + r) * HWSZ + lr]      = acc10[r];
    o0[(size_t)(mg + 16 + r) * HWSZ + 16 + lr] = acc11[r];
  }
}

// ---------------------------------------------------------------------------
extern "C" void kernel_launch(void* const* d_in, const int* in_sizes, int n_in,
                              void* d_out, int out_size, void* d_ws, size_t ws_size,
                              hipStream_t stream) {
  const float* x       = (const float*)d_in[0];
  const float* thetas  = (const float*)d_in[1];
  const float* scales  = (const float*)d_in[2];
  const float* lambdas = (const float*)d_in[3];
  const float* weight  = (const float*)d_in[4];
  float* out = (float*)d_out;

  // workspace: rot (f32, 64*81) at 0; bf16 transformed weights at +32KB
  float*  rot = (float*)d_ws;
  __bf16* gw  = (__bf16*)((char*)d_ws + 32768);

  rot_kernel<<<1, 64, 0, stream>>>(thetas, scales, lambdas, rot);
  wtrans_kernel<<<B_ * 256, 256, 0, stream>>>(weight, rot, gw);
  conv_wmma_kernel<<<dim3(HWSZ / 32, 1, B_), 256, 0, stream>>>(x, gw, out);
}
